// NeighborhoodAttention_68771016343757
// MI455X (gfx1250) — matrix operand order repaired and checked by reading, not compiled, not run on hardware
//
#include <hip/hip_runtime.h>

typedef _Float16 v16h __attribute__((ext_vector_type(16)));
typedef _Float16 v8h  __attribute__((ext_vector_type(8)));
typedef _Float16 v4h  __attribute__((ext_vector_type(4)));
typedef _Float16 h2   __attribute__((ext_vector_type(2)));
typedef float    v8f  __attribute__((ext_vector_type(8)));

#define T_     16
#define H_     32
#define W_     48
#define HEADS_ 16
#define D_     64
#define KT 5
#define KH 7
#define KW 7
#define ST 4
#define SH 4
#define SW 4
#define GT (T_/ST)        // 4
#define GH (H_/SH)        // 8
#define GW (W_/SW)        // 12
#define G_ (GT*GH*GW)     // 384
#define KWIN (KT*KH*KW)   // 245
#define PADK 256
#define QG 64
#define NEGINF (-3.0e38f)

__device__ __forceinline__ int group_start(int g, int n, int k, int s) {
    int lead = g * s + (s - 1) / 2;
    int st = lead - (k - 1) / 2;
    if (st < 0) st = 0;
    if (st > n - k) st = n - k;
    return st;
}

__device__ __forceinline__ v16h cat8(v8h lo, v8h hi) {
    return __builtin_shufflevector(lo, hi, 0,1,2,3,4,5,6,7,8,9,10,11,12,13,14,15);
}

// xor-16 lane exchange (SWAPX16) of a float / of 4 dwords holding 8 f16
__device__ __forceinline__ float swz16f(float a) {
    return __int_as_float(__builtin_amdgcn_ds_swizzle(__float_as_int(a), 0x401F));
}
__device__ __forceinline__ v8h swz16x8(v8h a) {
    union { v8h v; int i[4]; } u; u.v = a;
    #pragma unroll
    for (int t = 0; t < 4; ++t) u.i[t] = __builtin_amdgcn_ds_swizzle(u.i[t], 0x401F);
    return u.v;
}
__device__ __forceinline__ h2 pk2(float a, float b) {
    h2 r; r[0] = (_Float16)a; r[1] = (_Float16)b;   // fuses to v_cvt_pk_f16_f32
    return r;
}
union V8H { v8h v; h2 e[4]; };

// One 64-key transposed flash step: S^T = K*Q^T (4 key-tiles of 16 rows),
// softmax per lane (one query column per lane), O^T += V^T * P^T.
// FIRST: no correction needed. MASK: apply key>=KWIN mask (j compile-time).
template <bool FIRST, bool MASK>
__device__ __forceinline__ void gna_step64(
    int j, int n16, int half, int kk0,
    const v16h* bq, v16h ones,
    const _Float16* Ks, const _Float16* Vt,
    v8f* acc, float& rowmax, float& rowsum)
{
    // ---- S^T tiles: rows = keys (lane row n16), cols = queries ----
    v8f s[4];
    #pragma unroll
    for (int t = 0; t < 4; ++t) {
        v8f sz;
        #pragma unroll
        for (int r = 0; r < 8; ++r) sz[r] = 0.f;
        const int key = j + t * 16 + n16;            // this lane's A row
        #pragma unroll
        for (int c = 0; c < 2; ++c) {
            const v8h alo = *(const v8h*)&Ks[key * D_ + c * 32 + kk0];
            const v8h ahi = *(const v8h*)&Ks[key * D_ + c * 32 + kk0 + 8];
            sz = __builtin_amdgcn_wmma_f32_16x16x32_f16(
                    false, cat8(alo, ahi), false, bq[c], (short)0, sz, false, false);
        }
        s[t] = sz;
    }
    // lane's element (t, r) is key j + t*16 + r + 8*half of its query column
    if (MASK) {
        #pragma unroll
        for (int t = 0; t < 4; ++t)
            #pragma unroll
            for (int r = 0; r < 8; ++r) {
                const bool iv = (j + t * 16 + r + 8 * half) >= KWIN;  // folds for t<3
                s[t][r] = iv ? NEGINF : s[t][r];
            }
    }
    // ---- max over 64 keys: in-lane tree (v_max3) + one xor-16 exchange ----
    float m = s[0][0];
    #pragma unroll
    for (int t = 0; t < 4; ++t)
        #pragma unroll
        for (int r = 0; r < 8; ++r) m = fmaxf(m, s[t][r]);
    m = fmaxf(m, swz16f(m));
    const float nmax = FIRST ? m : fmaxf(rowmax, m);
    // ---- P^T = exp(S^T - nmax), packed f16 per tile ----
    v8h pk_[4];
    #pragma unroll
    for (int t = 0; t < 4; ++t) {
        V8H u;
        #pragma unroll
        for (int pr = 0; pr < 4; ++pr)
            u.e[pr] = pk2(__expf(s[t][2*pr] - nmax), __expf(s[t][2*pr+1] - nmax));
        pk_[t] = u.v;
    }
    // ---- build P^T B-operands in registers: own 8 keys + partner's 8 keys ----
    v16h bp[2];
    #pragma unroll
    for (int c = 0; c < 2; ++c) {
        const v8h send = half ? pk_[2*c] : pk_[2*c + 1];
        const v8h recv = swz16x8(send);
        bp[c] = cat8(half ? recv : pk_[2*c], half ? pk_[2*c + 1] : recv);
    }
    // ---- row sums via WMMA: ones * P^T -> every element is the column sum ----
    v8f rz;
    #pragma unroll
    for (int r = 0; r < 8; ++r) rz[r] = 0.f;
    rz = __builtin_amdgcn_wmma_f32_16x16x32_f16(false, ones, false, bp[0], (short)0, rz, false, false);
    rz = __builtin_amdgcn_wmma_f32_16x16x32_f16(false, ones, false, bp[1], (short)0, rz, false, false);
    const float rsum = rz[0];
    // ---- state update + O^T = O^T*corr + V^T * P^T ----
    if (FIRST) {
        rowsum = rsum; rowmax = nmax;
    } else {
        const float corr = __expf(rowmax - nmax);
        rowsum = rowsum * corr + rsum;
        rowmax = nmax;
        #pragma unroll
        for (int dt = 0; dt < 4; ++dt)
            #pragma unroll
            for (int r = 0; r < 8; ++r) acc[dt][r] *= corr;
    }
    #pragma unroll
    for (int dt = 0; dt < 4; ++dt) {
        const int dim = dt * 16 + n16;               // this lane's A row (V^T)
        v8f a = acc[dt];
        #pragma unroll
        for (int c = 0; c < 2; ++c) {
            const v8h alo = *(const v8h*)&Vt[dim * PADK + j + c * 32 + kk0];
            const v8h ahi = *(const v8h*)&Vt[dim * PADK + j + c * 32 + kk0 + 8];
            a = __builtin_amdgcn_wmma_f32_16x16x32_f16(
                    false, cat8(alo, ahi), false, bp[c], (short)0, a, false, false);
        }
        acc[dt] = a;
    }
}

__launch_bounds__(128)
__global__ void gna_fwd_kernel(const float* __restrict__ qp,
                               const float* __restrict__ kp,
                               const float* __restrict__ vp,
                               float* __restrict__ out)
{
    __shared__ _Float16 Qs[QG * D_];      // 8 KB, [q][d], pre-scaled
    __shared__ _Float16 Ks[PADK * D_];    // 32 KB, [key][d], zero padded
    __shared__ _Float16 Vt[D_ * PADK];    // 32 KB, [d][key], zero padded
    __shared__ int      KeyL[PADK];       // 1 KB key -> flat token index

    const int tid  = threadIdx.x;
    const int lane = tid & 31;
    const int wave = tid >> 5;

    const int g  = blockIdx.x / HEADS_;
    const int h  = blockIdx.x % HEADS_;
    const int gt = g / (GH * GW);
    const int gh = (g / GW) % GH;
    const int gw = g % GW;

    const int t0 = group_start(gt, T_, KT, ST);
    const int h0 = group_start(gh, H_, KH, SH);
    const int w0 = group_start(gw, W_, KW, SW);

    // ---- key -> flat token index table ----
    for (int kk = tid; kk < PADK; kk += 128) {
        int l = -1;
        if (kk < KWIN) {
            int it  = kk / (KH * KW);
            int rem = kk - it * (KH * KW);
            int ih = rem / KW, iw = rem - ih * KW;
            l = ((t0 + it) * H_ + (h0 + ih)) * W_ + (w0 + iw);
        }
        KeyL[kk] = l;
    }
    __syncthreads();

    // ---- stage Q (scaled by 1/sqrt(D) = 0.125) ----
    for (int c = tid; c < QG * (D_ / 4); c += 128) {
        int qi = c >> 4;
        int d4 = (c & 15) << 2;
        int qt = qi >> 4, qh = (qi >> 2) & 3, qw = qi & 3;
        int tt = gt * ST + qt, hh = gh * SH + qh, ww = gw * SW + qw;
        int lidx = (tt * H_ + hh) * W_ + ww;
        const float4 qv = *(const float4*)(qp + ((long)lidx * HEADS_ + h) * D_ + d4);
        v4h qh4 = { (_Float16)(qv.x * 0.125f), (_Float16)(qv.y * 0.125f),
                    (_Float16)(qv.z * 0.125f), (_Float16)(qv.w * 0.125f) };
        *(v4h*)&Qs[qi * D_ + d4] = qh4;
    }
    // ---- stage K row-major [key][d], zero padded ----
    for (int c = tid; c < PADK * (D_ / 4); c += 128) {
        int kk = c >> 4;
        int d4 = (c & 15) << 2;
        float4 kv = {0.f, 0.f, 0.f, 0.f};
        int l = KeyL[kk];
        if (l >= 0) kv = *(const float4*)(kp + ((long)l * HEADS_ + h) * D_ + d4);
        v4h kh4 = { (_Float16)kv.x, (_Float16)kv.y, (_Float16)kv.z, (_Float16)kv.w };
        *(v4h*)&Ks[kk * D_ + d4] = kh4;
    }
    // ---- stage V transposed [d][key]: 4 consecutive keys, one dim -> b64 store ----
    for (int c = tid; c < (PADK / 4) * D_; c += 128) {
        int d  = c & (D_ - 1);
        int kb = (c >> 6) << 2;          // key base, multiple of 4
        v4h vv;
        #pragma unroll
        for (int i = 0; i < 4; ++i) {
            int l = KeyL[kb + i];
            float f = (l >= 0) ? vp[((long)l * HEADS_ + h) * D_ + d] : 0.f;
            vv[i] = (_Float16)f;
        }
        *(v4h*)&Vt[d * PADK + kb] = vv;
    }
    __syncthreads();

    const int half = lane >> 4;          // 0 / 1
    const int n16  = lane & 15;
    const int kk0  = half ? 16 : 0;      // A-operand contiguous base
    const int klo  = half ? 16 : 0;      // B-operand contiguous base
    const int qbase = wave * 16;
    const int query = qbase + n16;       // this lane's query column

    // Preload Q^T B-operands (32d x 16q per d-chunk): contiguous 16 halves
    v16h bq[2];
    #pragma unroll
    for (int c = 0; c < 2; ++c) {
        const v8h lo = *(const v8h*)&Qs[query * D_ + c * 32 + klo];
        const v8h hi = *(const v8h*)&Qs[query * D_ + c * 32 + klo + 8];
        bq[c] = cat8(lo, hi);
    }
    v16h ones;
    #pragma unroll
    for (int i = 0; i < 16; ++i) ones[i] = (_Float16)1.0f;

    v8f acc[4];
    #pragma unroll
    for (int dt = 0; dt < 4; ++dt)
        #pragma unroll
        for (int r = 0; r < 8; ++r) acc[dt][r] = 0.f;
    float rowmax = NEGINF, rowsum = 0.f;

    // keys 0..63 (first, no correction), 64..191 (plain), 192..255 (masked tail)
    gna_step64<true,  false>(0,   n16, half, kk0, bq, ones, Ks, Vt, acc, rowmax, rowsum);
    gna_step64<false, false>(64,  n16, half, kk0, bq, ones, Ks, Vt, acc, rowmax, rowsum);
    gna_step64<false, false>(128, n16, half, kk0, bq, ones, Ks, Vt, acc, rowmax, rowsum);
    gna_step64<false, true >(192, n16, half, kk0, bq, ones, Ks, Vt, acc, rowmax, rowsum);

    // ---- normalize and store: 8 consecutive dims per tile -> b128 stores ----
    const float inv = 1.0f / rowsum;
    const int qt = query >> 4, qh2 = (query >> 2) & 3, qw2 = query & 3;
    const int tt = gt * ST + qt, hh = gh * SH + qh2, ww = gw * SW + qw2;
    const int lidx = (tt * H_ + hh) * W_ + ww;
    float* op = out + ((long)lidx * HEADS_ + h) * D_ + 8 * half;
    #pragma unroll
    for (int dt = 0; dt < 4; ++dt) {
        float4 f0 = { acc[dt][0] * inv, acc[dt][1] * inv,
                      acc[dt][2] * inv, acc[dt][3] * inv };
        float4 f1 = { acc[dt][4] * inv, acc[dt][5] * inv,
                      acc[dt][6] * inv, acc[dt][7] * inv };
        *(float4*)(op + dt * 16)     = f0;
        *(float4*)(op + dt * 16 + 4) = f1;
    }
}

extern "C" void kernel_launch(void* const* d_in, const int* in_sizes, int n_in,
                              void* d_out, int out_size, void* d_ws, size_t ws_size,
                              hipStream_t stream) {
    (void)in_sizes; (void)n_in; (void)out_size; (void)d_ws; (void)ws_size;
    const float* q = (const float*)d_in[0];
    const float* k = (const float*)d_in[1];
    const float* v = (const float*)d_in[2];
    float* out = (float*)d_out;
    gna_fwd_kernel<<<dim3(G_ * HEADS_), dim3(128), 0, stream>>>(q, k, v, out);
}
